// Net_85899345920617
// MI455X (gfx1250) — matrix operand order repaired and checked by reading, not compiled, not run on hardware
//
#include <hip/hip_runtime.h>
#include <hip/hip_bf16.h>
#include <math.h>

typedef __attribute__((ext_vector_type(2)))  float  v2f;
typedef __attribute__((ext_vector_type(8)))  float  v8f;
typedef __attribute__((ext_vector_type(16))) __bf16 v16bf;

#define GRIDN   400
#define NPIX    160000      // 400*400
#define SZ      200
#define PPAD    100
#define NB      64
#define T1D     25          // 400/16
#define MB      5           // M-tile register blocking (25 = 5*5, no edges)
#define WPI     125         // waves per image = (25/5) * 25
#define CCDN    160000      // 64*50*50

// ---------------------------------------------------------------------------
// DFT matrices with fftshift folded in:
//   F [k][n] = (-1)^k e^{-i 2pi kn/400}            (fft of fftshift'ed input)
//   G [m][k] = (1/400)(-1)^k e^{+i 2pi mk/400}     (ifftshift of ifft output)
// plus explicit transposes so every matmul is plain row-major A*B.
// ---------------------------------------------------------------------------
__global__ void init_dft(float* F, float* Fi, float* G, float* Gi,
                         float* FT, float* FTi, float* GT, float* GTi)
{
    int t = blockIdx.x * blockDim.x + threadIdx.x;
    if (t >= NPIX) return;
    int a = t / GRIDN, b = t % GRIDN;
    const float w = 6.283185307179586476925f / GRIDN;
    int m = (a * b) % GRIDN;                 // exact integer phase reduction
    float s, c;
    sincosf(w * (float)m, &s, &c);
    float sgA = (a & 1) ? -1.f : 1.f;
    float sgB = (b & 1) ? -1.f : 1.f;
    const float inv = 1.0f / GRIDN;
    F [t] =  c * sgA;        Fi [t] = -s * sgA;       // F[a][b], shift sign on row a
    FT[t] =  c * sgB;        FTi[t] = -s * sgB;       // FT[a][b] = F[b][a]
    G [t] =  c * sgB * inv;  Gi [t] =  s * sgB * inv; // G[a][b], shift sign on col b
    GT[t] =  c * sgA * inv;  GTi[t] =  s * sgA * inv; // GT[a][b] = G[b][a]
}

// pad (B,200,200) -> (B,400,400); sim==nullptr means purely real source (x)
__global__ void pad_wave(const float* __restrict__ sre, const float* __restrict__ sim,
                         float* __restrict__ dre, float* __restrict__ dim_, int nb)
{
    long t = (long)blockIdx.x * blockDim.x + threadIdx.x;
    if (t >= (long)nb * NPIX) return;
    int b = (int)(t / NPIX), p = (int)(t % NPIX);
    int r = p / GRIDN, c = p % GRIDN;
    float re = 0.f, im = 0.f;
    if (r >= PPAD && r < PPAD + SZ && c >= PPAD && c < PPAD + SZ) {
        long si = (long)b * SZ * SZ + (long)(r - PPAD) * SZ + (c - PPAD);
        re = sre[si];
        if (sim) im = sim[si];
    }
    dre[t] = re; dim_[t] = im;
}

// ---------------------------------------------------------------------------
// Batched complex 400x400x400 GEMM on V_WMMA_F32_16X16X4_F32.
// One wave32 owns FIVE 16x16 complex output tiles stacked along M (25 = 5*5,
// so no edge guards and EXEC stays all-ones).  The B fragment is loaded once
// per K-step and reused by all 5 accumulator pairs -> 20 WMMA per 96 B/lane.
// sA/sB==0 selects the shared DFT matrix; otherwise per-image stride NPIX.
// Optional fused multiply by a complex64 transfer function (float2 pairs).
// ---------------------------------------------------------------------------
__global__ void __launch_bounds__(256)
cgemm400_m5(float* __restrict__ Cre, float* __restrict__ Cim,
            const float* __restrict__ Are, const float* __restrict__ Aim, long sA,
            const float* __restrict__ Bre, const float* __restrict__ Bim, long sB,
            const float2* __restrict__ trans, int totalWaves)
{
    int wave = (int)((blockIdx.x * blockDim.x + threadIdx.x) >> 5);
    if (wave >= totalWaves) return;
    int lane = threadIdx.x & 31;
    int hi = lane >> 4, l15 = lane & 15;
    int img = wave / WPI;
    int rem = wave % WPI;
    int bm  = rem / T1D;                 // M-block 0..4 -> tiles bm*5 .. bm*5+4
    int tn  = rem % T1D;

    const float* abase_r = Are + (long)img * sA;
    const float* abase_i = Aim + (long)img * sA;
    // B fragment base: row = k0 + hi*2 + {0,1}, col = tn*16+l15 (stride-400 pair)
    const float* br = Bre + (long)img * sB + (long)(hi * 2) * GRIDN + tn * 16 + l15;
    const float* bi = Bim + (long)img * sB + (long)(hi * 2) * GRIDN + tn * 16 + l15;

    v8f cre[MB] = {};
    v8f cim[MB] = {};

    for (int k0 = 0; k0 < GRIDN; k0 += 4) {
        const float* brk = br + (long)k0 * GRIDN;
        const float* bik = bi + (long)k0 * GRIDN;
        if ((k0 & 31) == 0 && k0 + 32 < GRIDN) {       // stream next K-block of B
            __builtin_prefetch(brk + 32 * GRIDN, 0, 1);  // -> global_prefetch_b8
            __builtin_prefetch(bik + 32 * GRIDN, 0, 1);
        }
        v2f Br, Bi;
        Br[0] = brk[0];  Br[1] = brk[GRIDN];
        Bi[0] = bik[0];  Bi[1] = bik[GRIDN];
#pragma unroll
        for (int t = 0; t < MB; ++t) {
            // A fragment: row = (bm*5+t)*16+l15, k = k0 + hi*2 + {0,1} (contiguous)
            const float* ar = abase_r + (long)((bm * MB + t) * 16 + l15) * GRIDN + k0 + hi * 2;
            const float* ai = abase_i + (long)((bm * MB + t) * 16 + l15) * GRIDN + k0 + hi * 2;
            v2f Ar, Ai;
            Ar[0] = ar[0];  Ar[1] = ar[1];
            Ai[0] = ai[0];  Ai[1] = ai[1];
            v2f nAi = -Ai;  // f32 WMMA has no A/B NEG bits (ISA NEG table) -> negate fragment
            cre[t] = __builtin_amdgcn_wmma_f32_16x16x4_f32(false, Ar,  false, Br, (short)0, cre[t], false, false);
            cre[t] = __builtin_amdgcn_wmma_f32_16x16x4_f32(false, nAi, false, Bi, (short)0, cre[t], false, false);
            cim[t] = __builtin_amdgcn_wmma_f32_16x16x4_f32(false, Ar,  false, Bi, (short)0, cim[t], false, false);
            cim[t] = __builtin_amdgcn_wmma_f32_16x16x4_f32(false, Ai,  false, Br, (short)0, cim[t], false, false);
        }
    }

    long cbase = (long)img * NPIX;
#pragma unroll
    for (int t = 0; t < MB; ++t) {
#pragma unroll
        for (int j = 0; j < 8; ++j) {
            int row = (bm * MB + t) * 16 + hi * 8 + j;  // C/D layout: VGPR j -> M=j (+8 hi)
            int col = tn * 16 + l15;
            long idx = cbase + (long)row * GRIDN + col;
            float re = cre[t][j], im = cim[t][j];
            if (trans) {
                float2 tv = trans[(long)row * GRIDN + col];
                float nr = re * tv.x - im * tv.y;
                float ni = re * tv.y + im * tv.x;
                re = nr; im = ni;
            }
            Cre[idx] = re; Cim[idx] = im;
        }
    }
}

// crop (B,400,400) -> (B,200,200), optionally * exp(i*11*pi*(sin(2*phi)+1))
__global__ void crop_phase(const float* __restrict__ sre, const float* __restrict__ sim,
                           float* __restrict__ wre, float* __restrict__ wim,
                           const float* __restrict__ phase, int nb)
{
    long t = (long)blockIdx.x * blockDim.x + threadIdx.x;
    if (t >= (long)nb * SZ * SZ) return;
    int b = (int)(t / (SZ * SZ)), p = (int)(t % (SZ * SZ));
    int r = p / SZ, c = p % SZ;
    long si = (long)b * NPIX + (long)(r + PPAD) * GRIDN + (c + PPAD);
    float re = sre[si], im = sim[si];
    if (phase) {
        float ph = phase[p];
        float ang = 34.557519189487726f * (sinf(2.0f * ph) + 1.0f);  // 11*pi*(sin(2x)+1)
        float s, cc;
        sincosf(ang, &s, &cc);
        float nr = re * cc - im * s;
        float ni = re * s + im * cc;
        re = nr; im = ni;
    }
    wre[t] = re; wim[t] = im;
}

// |w|^2 on cropped region + 4x4 average pool -> (B,50,50)
__global__ void intensity_pool(const float* __restrict__ sre, const float* __restrict__ sim,
                               float* __restrict__ ccd, int nb)
{
    int t = blockIdx.x * blockDim.x + threadIdx.x;
    if (t >= nb * 2500) return;
    int b = t / 2500, p = t % 2500;
    int i = p / 50, j = p % 50;
    float acc = 0.f;
#pragma unroll
    for (int u = 0; u < 4; ++u)
#pragma unroll
        for (int v = 0; v < 4; ++v) {
            long si = (long)b * NPIX + (long)(PPAD + i * 4 + u) * GRIDN + (PPAD + j * 4 + v);
            float re = sre[si], im = sim[si];
            acc += re * re + im * im;
        }
    ccd[t] = acc * (1.0f / 16.0f);
}

// deterministic two-stage sum / sumsq reduction (128 fixed blocks)
__global__ void reduce_stats(const float* __restrict__ x, float* __restrict__ partial, int n)
{
    __shared__ float sh[256], sh2[256];
    float a = 0.f, b = 0.f;
    for (int i = blockIdx.x * blockDim.x + threadIdx.x; i < n; i += gridDim.x * blockDim.x) {
        float v = x[i]; a += v; b += v * v;
    }
    sh[threadIdx.x] = a; sh2[threadIdx.x] = b;
    __syncthreads();
    for (int st = 128; st > 0; st >>= 1) {
        if ((int)threadIdx.x < st) { sh[threadIdx.x] += sh[threadIdx.x + st]; sh2[threadIdx.x] += sh2[threadIdx.x + st]; }
        __syncthreads();
    }
    if (threadIdx.x == 0) { partial[blockIdx.x * 2] = sh[0]; partial[blockIdx.x * 2 + 1] = sh2[0]; }
}

__global__ void finalize_stats(const float* __restrict__ partial, float* __restrict__ stats, int nblk)
{
    if (threadIdx.x == 0 && blockIdx.x == 0) {
        float a = 0.f, b = 0.f;
        for (int i = 0; i < nblk; ++i) { a += partial[i * 2]; b += partial[i * 2 + 1]; }
        stats[0] = a; stats[1] = b;
    }
}

// batchnorm (global mean/var) -> bf16 features
__global__ void bn_feat(const float* __restrict__ ccd, const float* __restrict__ stats,
                        const float* __restrict__ gamma, const float* __restrict__ beta,
                        __bf16* __restrict__ feat, int n)
{
    int t = blockIdx.x * blockDim.x + threadIdx.x;
    if (t >= n) return;
    float mu  = stats[0] / (float)n;
    float var = stats[1] / (float)n - mu * mu;
    float v = gamma[0] * (ccd[t] - mu) * rsqrtf(var + 1e-5f) + beta[0];
    feat[t] = (__bf16)v;
}

__global__ void f2bf(const float* __restrict__ src, __bf16* __restrict__ dst, int n)
{
    for (int i = blockIdx.x * blockDim.x + threadIdx.x; i < n; i += gridDim.x * blockDim.x)
        dst[i] = (__bf16)src[i];
}

// ---------------------------------------------------------------------------
// C[M,N] = act( (A[M,K] * B[N,K]^T + bias) * scale )  on V_WMMA_F32_16X16X32_BF16
// TN layout -> per-lane contiguous K for both fragments; guarded zero-fill keeps
// EXEC all-ones (WMMA requirement) for the K/N edges (2500, 21).
// ---------------------------------------------------------------------------
__global__ void __launch_bounds__(256)
gemm_bf16_tn(float* __restrict__ Cf, __bf16* __restrict__ Cbf,
             const __bf16* __restrict__ A, const __bf16* __restrict__ B,
             const float* __restrict__ bias, int M, int N, int K,
             float scale, int relu)
{
    int wave = (int)((blockIdx.x * blockDim.x + threadIdx.x) >> 5);
    int tilesN = (N + 15) / 16;
    int tilesM = M / 16;                       // M == 64 here
    if (wave >= tilesM * tilesN) return;
    int lane = threadIdx.x & 31;
    int hi = lane >> 4, l15 = lane & 15;
    int tm = wave / tilesN, tn = wave % tilesN;
    int m = tm * 16 + l15;                     // always < M
    int n = tn * 16 + l15;

    v8f acc = {};
    for (int k0 = 0; k0 < K; k0 += 32) {
        v16bf a, b;
#pragma unroll
        for (int e = 0; e < 8; ++e) {          // A 16x32: elems 0..15 -> k0+hi*16+e (contiguous)
            int k1 = k0 + hi * 16 + e;
            int k2 = k0 + hi * 16 + 8 + e;
            a[e]     = (k1 < K) ? A[(long)m * K + k1] : (__bf16)0.f;
            a[e + 8] = (k2 < K) ? A[(long)m * K + k2] : (__bf16)0.f;
        }
#pragma unroll
        for (int e = 0; e < 16; ++e) {         // B 32x16 (from W^T): k = k0 + hi*16 + e
            int k = k0 + hi * 16 + e;
            b[e] = (n < N && k < K) ? B[(long)n * K + k] : (__bf16)0.f;
        }
        acc = __builtin_amdgcn_wmma_f32_16x16x32_bf16(false, a, false, b, (short)0, acc, false, false);
    }
#pragma unroll
    for (int j = 0; j < 8; ++j) {
        int row = tm * 16 + hi * 8 + j;
        int col = tn * 16 + l15;
        if (col < N) {
            float v = (acc[j] + bias[col]) * scale;
            if (relu) v = fmaxf(v, 0.f);
            if (Cf)  Cf [(long)row * N + col] = v;
            if (Cbf) Cbf[(long)row * N + col] = (__bf16)v;
        }
    }
}

__global__ void softmax_rows(const float* __restrict__ logits, float* __restrict__ out, int nb, int nc)
{
    int b = blockIdx.x * blockDim.x + threadIdx.x;
    if (b >= nb) return;
    const float* L = logits + (long)b * nc;
    float mx = L[0];
    for (int i = 1; i < nc; ++i) mx = fmaxf(mx, L[i]);
    float s = 0.f;
    for (int i = 0; i < nc; ++i) s += expf(L[i] - mx);
    float inv = 1.0f / s;
    for (int i = 0; i < nc; ++i) out[(long)b * nc + i] = expf(L[i] - mx) * inv;
}

// ---------------------------------------------------------------------------
extern "C" void kernel_launch(void* const* d_in, const int* in_sizes, int n_in,
                              void* d_out, int out_size, void* d_ws, size_t ws_size,
                              hipStream_t stream)
{
    (void)in_sizes; (void)n_in; (void)out_size; (void)ws_size;
    const float*  x     = (const float*) d_in[0];   // (64,200,200)
    const float*  phase = (const float*) d_in[1];   // (5,200,200)
    const float2* h_pro = (const float2*)d_in[2];   // (400,400) complex64
    const float2* h     = (const float2*)d_in[3];
    const float2* h_det = (const float2*)d_in[4];
    const float*  W1    = (const float*) d_in[5];   // (2500,2500)
    const float*  b1    = (const float*) d_in[6];
    const float*  W2    = (const float*) d_in[7];   // (21,2500)
    const float*  b2    = (const float*) d_in[8];
    const float*  gamma = (const float*) d_in[9];
    const float*  beta  = (const float*) d_in[10];

    float* w = (float*)d_ws;
    const long MAT = NPIX;
    float* F   = w;              float* Fi  = w + 1 * MAT;
    float* G   = w + 2 * MAT;    float* Gi  = w + 3 * MAT;
    float* FT  = w + 4 * MAT;    float* FTi = w + 5 * MAT;
    float* GT  = w + 6 * MAT;    float* GTi = w + 7 * MAT;
    float* waveR = w + 8 * MAT;                 // 64*200*200
    float* waveI = waveR + (long)NB * SZ * SZ;
    float* AR = waveI + (long)NB * SZ * SZ;     // 64*400*400 each
    float* AI = AR + (long)NB * NPIX;
    float* BR = AI + (long)NB * NPIX;
    float* BI = BR + (long)NB * NPIX;
    float* ccd     = BI + (long)NB * NPIX;      // 160000
    float* partial = ccd + CCDN;                // 256
    float* stats   = partial + 256;             // 2
    float* logits  = stats + 8;                 // 64*21
    __bf16* featbf = (__bf16*)(logits + 2048);  // 160000 bf16
    __bf16* hidbf  = featbf + CCDN;             // 160000 bf16
    __bf16* W1bf   = hidbf + CCDN;              // 6,250,000 bf16
    __bf16* W2bf   = W1bf + 6250000L;           // 52,500 bf16

    const int TW = NB * WPI;                    // 8000 waves per cgemm pass
    const int GEMM_BLOCKS = TW / 8;             // 8 waves / 256-thread block
    const int PADT = (NB * NPIX + 255) / 256;
    const int CRPT = (NB * SZ * SZ + 255) / 256;

    init_dft<<<(NPIX + 255) / 256, 256, 0, stream>>>(F, Fi, G, Gi, FT, FTi, GT, GTi);

    auto diffract = [&](const float2* trans) {
        // FFT2 (shift folded): B = F*A ; A = B*FT  (x transfer fused)
        cgemm400_m5<<<GEMM_BLOCKS, 256, 0, stream>>>(BR, BI, F, Fi, 0, AR, AI, NPIX, nullptr, TW);
        cgemm400_m5<<<GEMM_BLOCKS, 256, 0, stream>>>(AR, AI, BR, BI, NPIX, FT, FTi, 0, trans, TW);
        // IFFT2 (+ifftshift, 1/N^2 folded): B = G*A ; A = B*GT
        cgemm400_m5<<<GEMM_BLOCKS, 256, 0, stream>>>(BR, BI, G, Gi, 0, AR, AI, NPIX, nullptr, TW);
        cgemm400_m5<<<GEMM_BLOCKS, 256, 0, stream>>>(AR, AI, BR, BI, NPIX, GT, GTi, 0, nullptr, TW);
    };

    // input diffraction through h_pro
    pad_wave<<<PADT, 256, 0, stream>>>(x, nullptr, AR, AI, NB);
    diffract(h_pro);
    crop_phase<<<CRPT, 256, 0, stream>>>(AR, AI, waveR, waveI, nullptr, NB);

    // 5 phase layers through h
    for (int l = 0; l < 5; ++l) {
        pad_wave<<<PADT, 256, 0, stream>>>(waveR, waveI, AR, AI, NB);
        diffract(h);
        crop_phase<<<CRPT, 256, 0, stream>>>(AR, AI, waveR, waveI, phase + (long)l * SZ * SZ, NB);
    }

    // detector diffraction through h_det, intensity + 4x4 pool
    pad_wave<<<PADT, 256, 0, stream>>>(waveR, waveI, AR, AI, NB);
    diffract(h_det);
    intensity_pool<<<(CCDN + 255) / 256, 256, 0, stream>>>(AR, AI, ccd, NB);

    // deterministic batchnorm stats
    reduce_stats<<<128, 256, 0, stream>>>(ccd, partial, CCDN);
    finalize_stats<<<1, 32, 0, stream>>>(partial, stats, 128);
    bn_feat<<<(CCDN + 255) / 256, 256, 0, stream>>>(ccd, stats, gamma, beta, featbf, CCDN);

    // MLP on bf16 WMMA
    f2bf<<<512, 256, 0, stream>>>(W1, W1bf, 6250000);
    f2bf<<<64, 256, 0, stream>>>(W2, W2bf, 52500);

    {   // hidden = relu(feat @ W1^T + b1): M=64, N=K=2500 -> 4*157 tiles
        int waves = 4 * ((2500 + 15) / 16);
        gemm_bf16_tn<<<(waves + 7) / 8, 256, 0, stream>>>(nullptr, hidbf, featbf, W1bf, b1,
                                                          64, 2500, 2500, 1.0f, 1);
    }
    {   // logits = (hidden @ W2^T + b2) / T: M=64, N=21, K=2500 -> 8 tiles
        gemm_bf16_tn<<<1, 256, 0, stream>>>(logits, nullptr, hidbf, W2bf, b2,
                                            64, 21, 2500, 10.0f, 0);
    }
    softmax_rows<<<2, 32, 0, stream>>>(logits, (float*)d_out, NB, 21);
}
